// EdgeDecoder_43885975831150
// MI455X (gfx1250) — matrix operand order repaired and checked by reading, not compiled, not run on hardware
//
#include <hip/hip_runtime.h>
#include <hip/hip_bf16.h>
#include <math.h>

typedef _Float16 v16h __attribute__((ext_vector_type(16)));
typedef _Float16 v8h  __attribute__((ext_vector_type(8)));
typedef float    v8f  __attribute__((ext_vector_type(8)));

#define EMB   128
#define HID   256
#define NOUT  10
#define MBLK  32                  // edges per workgroup (2 M-tiles of 16)
#define LDA   264                 // padded LDS row stride in halves (256 + 8 -> 4-bank rotate/row)

// f16 weight scratch layout in d_ws (units: halves)
#define W1H_OFF 0
#define W2H_OFF (256 * 256)
#define W3H_OFF (2 * 256 * 256)
#define W3ROWS  16                // W3 zero-padded 10 -> 16 rows
#define WS_HALFS (2 * 256 * 256 + W3ROWS * 256)

// ---------------------------------------------------------------------------
// Convert fp32 weights to f16 once per launch (W3 zero-padded to 16 rows).
// ---------------------------------------------------------------------------
__global__ void prep_weights_f16(const float* __restrict__ W1,
                                 const float* __restrict__ W2,
                                 const float* __restrict__ W3,
                                 _Float16* __restrict__ wsh) {
  int i = blockIdx.x * blockDim.x + threadIdx.x;
  if (i < 256 * 256) {
    wsh[W1H_OFF + i] = (_Float16)W1[i];
  } else if (i < 2 * 256 * 256) {
    wsh[i] = (_Float16)W2[i - 256 * 256];
  } else if (i < WS_HALFS) {
    int j = i - 2 * 256 * 256;
    int r = j >> 8, c = j & 255;
    wsh[i] = (r < NOUT) ? (_Float16)W3[r * 256 + c] : (_Float16)0.0f;
  }
}

__device__ __forceinline__ v16h cat8(v8h lo, v8h hi) {
  return __builtin_shufflevector(lo, hi,
           0,1,2,3,4,5,6,7,8,9,10,11,12,13,14,15);
}

// ---------------------------------------------------------------------------
// One layer: D(32 edges x 256) = relu(A(32x256) @ W^T + b), f16 in / f32 acc.
// 2x2 register blocking: each wave owns 2 N-tiles x 2 M-tiles. Inner K step
// loads {a0,a1,b0,b1} once and issues 4 WMMAs -> 1 ds_b128 + 1 global_b128
// per v_wmma. A-fragment per ISA 16-bit A 16x32 layout (lane half -> K+0/K+8,
// chunks [k,k+8)+[k+16,k+24)); B column n == row n of W, same K chunks as A.
// ---------------------------------------------------------------------------
__device__ __forceinline__ void mlp_layer_wmma(const _Float16* __restrict__ Ain,
                                               _Float16* __restrict__ Aout,
                                               const _Float16* __restrict__ Wh,
                                               const float* __restrict__ bias,
                                               int wave, int lane) {
  const int lhalf = lane >> 4;     // 0: lanes 0-15, 1: lanes 16-31
  const int lm    = lane & 15;
  const int n0 = (wave * 2 + 0) * 16 + lm;   // lane's column in N-tile 0
  const int n1 = (wave * 2 + 1) * 16 + lm;   // lane's column in N-tile 1

  const _Float16* __restrict__ w0 = Wh + (size_t)n0 * 256;
  const _Float16* __restrict__ w1 = Wh + (size_t)n1 * 256;
  const _Float16* __restrict__ ar0 = Ain + lm * LDA;          // M-tile 0
  const _Float16* __restrict__ ar1 = Ain + (16 + lm) * LDA;   // M-tile 1

  v8f acc00 = {0.f,0.f,0.f,0.f,0.f,0.f,0.f,0.f};
  v8f acc01 = acc00, acc10 = acc00, acc11 = acc00;

#pragma unroll
  for (int kt = 0; kt < 8; ++kt) {
    const int ka = kt * 32 + lhalf * 8;    // A frag chunk base
    const int kb = kt * 32 + lhalf * 8;    // B frag chunk base
    v16h a0 = cat8(*(const v8h*)&ar0[ka], *(const v8h*)&ar0[ka + 16]);
    v16h a1 = cat8(*(const v8h*)&ar1[ka], *(const v8h*)&ar1[ka + 16]);
    v16h b0 = cat8(*(const v8h*)(w0 + kb), *(const v8h*)(w0 + kb + 16));
    v16h b1 = cat8(*(const v8h*)(w1 + kb), *(const v8h*)(w1 + kb + 16));
    acc00 = __builtin_amdgcn_wmma_f32_16x16x32_f16(false, a0, false, b0,
                                                   (short)0, acc00, false, false);
    acc01 = __builtin_amdgcn_wmma_f32_16x16x32_f16(false, a0, false, b1,
                                                   (short)0, acc01, false, false);
    acc10 = __builtin_amdgcn_wmma_f32_16x16x32_f16(false, a1, false, b0,
                                                   (short)0, acc10, false, false);
    acc11 = __builtin_amdgcn_wmma_f32_16x16x32_f16(false, a1, false, b1,
                                                   (short)0, acc11, false, false);
  }

  const float bv0 = bias[n0];
  const float bv1 = bias[n1];
#pragma unroll
  for (int r = 0; r < 8; ++r) {          // C layout: VGPR r -> M = r + 8*lhalf
    const int mr = r + lhalf * 8;
    float h;
    h = acc00[r] + bv0; h = h > 0.f ? h : 0.f;
    Aout[mr * LDA + n0] = (_Float16)h;
    h = acc01[r] + bv1; h = h > 0.f ? h : 0.f;
    Aout[mr * LDA + n1] = (_Float16)h;
    h = acc10[r] + bv0; h = h > 0.f ? h : 0.f;
    Aout[(16 + mr) * LDA + n0] = (_Float16)h;
    h = acc11[r] + bv1; h = h > 0.f ? h : 0.f;
    Aout[(16 + mr) * LDA + n1] = (_Float16)h;
  }
}

// ---------------------------------------------------------------------------
// Main kernel: one workgroup (256 thr / 8 waves) per 32-edge tile.
// ---------------------------------------------------------------------------
__global__ __launch_bounds__(256)
void edge_mlp_kernel(const float* __restrict__ zu, const float* __restrict__ zb,
                     const int* __restrict__ eidx,
                     const _Float16* __restrict__ wsh,
                     const float* __restrict__ b1, const float* __restrict__ b2,
                     const float* __restrict__ b3,
                     float* __restrict__ out, int E) {
  __shared__ _Float16 Abuf[MBLK * LDA];
  __shared__ _Float16 Hbuf[MBLK * LDA];
  __shared__ float    Cl[MBLK][12];

  const int tid  = threadIdx.x;
  const int lane = tid & 31;
  const int wave = tid >> 5;
  const int e0   = blockIdx.x * MBLK;

  // ---- Phase 0: gather z_user[row] ++ z_book[col] -> f16 LDS tile [32 x 256]
#pragma unroll
  for (int p = tid; p < MBLK * 16; p += 256) {
    const int m = p >> 4;               // edge within tile
    const int c = p & 15;               // 16-column chunk
    int e = e0 + m; e = (e < E) ? e : (E - 1);
    const float* base = (c < 8)
        ? (zu + (size_t)eidx[e]     * EMB + c * 16)
        : (zb + (size_t)eidx[E + e] * EMB + (c - 8) * 16);
    float4 f0 = ((const float4*)base)[0];
    float4 f1 = ((const float4*)base)[1];
    float4 f2 = ((const float4*)base)[2];
    float4 f3 = ((const float4*)base)[3];
    v8h h0 = { (_Float16)f0.x, (_Float16)f0.y, (_Float16)f0.z, (_Float16)f0.w,
               (_Float16)f1.x, (_Float16)f1.y, (_Float16)f1.z, (_Float16)f1.w };
    v8h h1 = { (_Float16)f2.x, (_Float16)f2.y, (_Float16)f2.z, (_Float16)f2.w,
               (_Float16)f3.x, (_Float16)f3.y, (_Float16)f3.z, (_Float16)f3.w };
    *(v8h*)&Abuf[m * LDA + c * 16]     = h0;
    *(v8h*)&Abuf[m * LDA + c * 16 + 8] = h1;
  }
  __syncthreads();

  // ---- Phase 1: h1 = relu(z @ W1^T + b1)      (WMMA, Abuf -> Hbuf)
  mlp_layer_wmma(Abuf, Hbuf, wsh + W1H_OFF, b1, wave, lane);
  __syncthreads();

  // ---- Phase 2: h2 = relu(h1 @ W2^T + b2)     (WMMA, Hbuf -> Abuf)
  mlp_layer_wmma(Hbuf, Abuf, wsh + W2H_OFF, b2, wave, lane);
  __syncthreads();

  // ---- Phase 3: logits = h2 @ W3p^T + b3 (WMMA; W3 zero-padded to 16 rows).
  // 32x16 logits tile = 2 M-tiles x 1 N-tile: waves 0 and 1 do 8 WMMAs each.
  // Wave-uniform branch keeps EXEC all-ones for the executing waves.
  if (wave < 2) {
    const int lhalf = lane >> 4;
    const int lm    = lane & 15;
    const int mt    = wave;                                  // M-tile index
    const _Float16* __restrict__ ar = Abuf + (mt * 16 + lm) * LDA;
    const _Float16* __restrict__ wr = wsh + W3H_OFF + (size_t)lm * 256;
    v8f acc = {0.f,0.f,0.f,0.f,0.f,0.f,0.f,0.f};
#pragma unroll
    for (int kt = 0; kt < 8; ++kt) {
      const int ka = kt * 32 + lhalf * 8;
      const int kb = kt * 32 + lhalf * 8;
      v16h a = cat8(*(const v8h*)&ar[ka], *(const v8h*)&ar[ka + 16]);
      v16h b = cat8(*(const v8h*)(wr + kb), *(const v8h*)(wr + kb + 16));
      acc = __builtin_amdgcn_wmma_f32_16x16x32_f16(false, a, false, b,
                                                   (short)0, acc, false, false);
    }
    if (lm < NOUT) {                    // columns 10..15 are zero-pad, drop
      const float bv = b3[lm];
#pragma unroll
      for (int r = 0; r < 8; ++r)
        Cl[mt * 16 + r + lhalf * 8][lm] = acc[r] + bv;
    }
  }
  __syncthreads();

  // ---- Phase 4: log_softmax over 10 classes, write [E,10] fp32
  if (tid < MBLK) {
    const int e = e0 + tid;
    if (e < E) {
      float mx = Cl[tid][0];
#pragma unroll
      for (int o = 1; o < NOUT; ++o) mx = fmaxf(mx, Cl[tid][o]);
      float s = 0.f;
#pragma unroll
      for (int o = 0; o < NOUT; ++o) s += expf(Cl[tid][o] - mx);
      const float lse = logf(s) + mx;
      float* op = out + (size_t)e * NOUT;
#pragma unroll
      for (int o = 0; o < NOUT; ++o) op[o] = Cl[tid][o] - lse;
    }
  }
}

// ---------------------------------------------------------------------------
extern "C" void kernel_launch(void* const* d_in, const int* in_sizes, int n_in,
                              void* d_out, int out_size, void* d_ws, size_t ws_size,
                              hipStream_t stream) {
  const float* zu  = (const float*)d_in[0];   // z_user  [N,128] f32
  const float* zb  = (const float*)d_in[1];   // z_book  [N,128] f32
  const int*  eidx = (const int*)d_in[2];     // edge_label_index [2,E] int32
  const float* W1  = (const float*)d_in[3];   // [256,256]
  const float* b1  = (const float*)d_in[4];   // [256]
  const float* W2  = (const float*)d_in[5];   // [256,256]
  const float* b2  = (const float*)d_in[6];   // [256]
  const float* W3  = (const float*)d_in[7];   // [10,256]
  const float* b3  = (const float*)d_in[8];   // [10]
  float* out = (float*)d_out;
  const int E = in_sizes[2] / 2;

  _Float16* wsh = (_Float16*)d_ws;
  prep_weights_f16<<<(WS_HALFS + 255) / 256, 256, 0, stream>>>(W1, W2, W3, wsh);

  const int nblk = (E + MBLK - 1) / MBLK;
  edge_mlp_kernel<<<nblk, 256, 0, stream>>>(zu, zb, eidx, wsh,
                                            b1, b2, b3, out, E);
}
